// MultiHeadAttentionQuantum_65481071396905
// MI455X (gfx1250) — matrix operand order repaired and checked by reading, not compiled, not run on hardware
//
#include <hip/hip_runtime.h>
#include <hip/hip_bf16.h>

typedef __attribute__((ext_vector_type(16))) _Float16 v16h;
typedef __attribute__((ext_vector_type(8)))  _Float16 v8h;
typedef __attribute__((ext_vector_type(8)))  float    v8f;

#define WMMA_F16(a, b, c) \
  __builtin_amdgcn_wmma_f32_16x16x32_f16(false, (a), false, (b), (short)0, (c), false, false)

// ---------------------------------------------------------------------------
// Fragment loaders (wave32, V_WMMA_F32_16X16X32_F16 layouts per CDNA5 ISA 7.12.2)
// ---------------------------------------------------------------------------

// A-matrix 16x32 f16 from row-major storage (ld halves per row).
// lanes 0-15: M=lane, K = {0..7, 16..23}; lanes 16-31: M=lane-16, K = {8..15, 24..31}.
__device__ inline v16h load_a_rm(const _Float16* base, int ld, int lane) {
  const int row = lane & 15;
  const int kb  = (lane >> 4) << 3;
  const _Float16* p = base + row * ld + kb;
  v8h lo = *(const v8h*)(p);
  v8h hi = *(const v8h*)(p + 16);
  v16h a;
#pragma unroll
  for (int i = 0; i < 8; ++i) { a[i] = lo[i]; a[8 + i] = hi[i]; }
  return a;
}

// B-matrix 32x16 f16 from "N-major, K-contiguous" storage:
// element B[k][n] at base[n*ld + k]. lanes 0-15: K = {0..7, 16..23} (N=lane);
// lanes 16-31: K = {8..15, 24..31} (N=lane-16).
__device__ inline v16h load_b_kcont(const _Float16* base, int ld, int lane, int kc) {
  const int col = lane & 15;
  const int kh  = (lane >> 4) << 3;
  const _Float16* p = base + col * ld + kc + kh;
  v8h lo = *(const v8h*)(p);
  v8h hi = *(const v8h*)(p + 16);
  v16h b;
#pragma unroll
  for (int i = 0; i < 8; ++i) { b[i] = lo[i]; b[8 + i] = hi[i]; }
  return b;
}

// C/D f32 16x16: VGPR r holds (row = r + 8*(lane>=16), col = lane&15).
__device__ inline int cd_row(int r, int lane) { return r + ((lane >> 4) << 3); }

// ---------------------------------------------------------------------------
// Weight conversion: f32 row-major W[k][n] -> f16 transposed Wt[n][k]
// ---------------------------------------------------------------------------
__global__ __launch_bounds__(256) void convert_wo_kernel(const float* __restrict__ Wo,
                                                         _Float16* __restrict__ WoT) {
  int e = blockIdx.x * 256 + threadIdx.x;      // 0 .. 1024*1024-1
  int n = e >> 10, k = e & 1023;
  WoT[e] = (_Float16)Wo[(size_t)k * 1024 + n];
}

__global__ __launch_bounds__(256) void convert_w64_kernel(
    const float* __restrict__ Wq, const float* __restrict__ Wk, const float* __restrict__ Wv,
    _Float16* __restrict__ WtQ, _Float16* __restrict__ WtK, _Float16* __restrict__ WtV) {
  int e = blockIdx.x * 256 + threadIdx.x;      // 0 .. 4095
  if (e < 4096) {
    int n = e >> 6, k = e & 63;
    WtQ[e] = (_Float16)Wq[k * 64 + n];
    WtK[e] = (_Float16)Wk[k * 64 + n];
    WtV[e] = (_Float16)Wv[k * 64 + n];
  }
}

// ---------------------------------------------------------------------------
// Kernel 1: qp = cos(x+theta); q/k/v = qp @ W + b   (per (b,h), 64-token tiles)
// q,k stored f16 row-major (B,H,S,64); v stored f16 transposed (B,H,64,S).
// 128 threads = 4 waves, wave w owns token rows [w*16, w*16+16).
// WtQ|WtK|WtV (contiguous 24 KB) staged in LDS so every B-fragment is a ds load.
// ---------------------------------------------------------------------------
__global__ __launch_bounds__(128) void qkv_kernel(
    const float* __restrict__ x, const float* __restrict__ theta,
    const _Float16* __restrict__ WtQKV,   // WtQ followed by WtK, WtV (3*4096 halves)
    const float* __restrict__ bq, const float* __restrict__ bk, const float* __restrict__ bv,
    _Float16* __restrict__ qf, _Float16* __restrict__ kf, _Float16* __restrict__ vt) {
  __shared__ __align__(16) _Float16 sQP[64 * 64];   // 8 KB
  __shared__ __align__(16) _Float16 sW[3 * 64 * 64]; // 24 KB: WtQ, WtK, WtV

  const int tid = threadIdx.x, lane = tid & 31, wave = tid >> 5;
  const int bh = blockIdx.y;                 // b*16 + h
  const int st = blockIdx.x;                 // token tile (0..31)
  const int b = bh >> 4, h = bh & 15;

  // stage weights: 12288 halves, 128 threads x 6 x 16-half chunks
#pragma unroll
  for (int i = 0; i < 6; ++i) {
    const int e = (i * 128 + tid) * 16;      // 0..12272 step 16
    v8h w0 = *(const v8h*)(WtQKV + e);
    v8h w1 = *(const v8h*)(WtQKV + e + 8);
    *(v8h*)(&sW[e]) = w0;
    *(v8h*)(&sW[e + 8]) = w1;
  }

  // quantum feature map: cos(x + theta), tile (64 tokens x 64 dims) into LDS
#pragma unroll
  for (int i = 0; i < 32; ++i) {
    int e = i * 128 + tid;                   // 0..4095
    int r = e >> 6, d = e & 63;
    int s = st * 64 + r;
    float xv = x[((size_t)b * 2048 + s) * 1024 + h * 64 + d];
    sQP[e] = (_Float16)__cosf(xv + theta[d]);
  }
  __syncthreads();

  const _Float16* arow = &sQP[wave * 16 * 64];
  v16h a0 = load_a_rm(arow, 64, lane);        // K dims 0..31
  v16h a1 = load_a_rm(arow + 32, 64, lane);   // K dims 32..63
  const int col = lane & 15;

#pragma unroll
  for (int ns = 0; ns < 4; ++ns) {
    const _Float16* wq = &sW[0 * 4096 + ns * 16 * 64];
    const _Float16* wk = &sW[1 * 4096 + ns * 16 * 64];
    const _Float16* wv = &sW[2 * 4096 + ns * 16 * 64];
    v8f aq = {};
    aq = WMMA_F16(a0, load_b_kcont(wq, 64, lane, 0),  aq);
    aq = WMMA_F16(a1, load_b_kcont(wq, 64, lane, 32), aq);
    v8f ak = {};
    ak = WMMA_F16(a0, load_b_kcont(wk, 64, lane, 0),  ak);
    ak = WMMA_F16(a1, load_b_kcont(wk, 64, lane, 32), ak);
    v8f av = {};
    av = WMMA_F16(a0, load_b_kcont(wv, 64, lane, 0),  av);
    av = WMMA_F16(a1, load_b_kcont(wv, 64, lane, 32), av);

    const int n = ns * 16 + col;
    const float biq = bq[n], bik = bk[n], biv = bv[n];
#pragma unroll
    for (int r = 0; r < 8; ++r) {
      const int s = st * 64 + wave * 16 + cd_row(r, lane);
      qf[((size_t)bh * 2048 + s) * 64 + n] = (_Float16)(aq[r] + biq);
      kf[((size_t)bh * 2048 + s) * 64 + n] = (_Float16)(ak[r] + bik);
      vt[(size_t)bh * 64 * 2048 + (size_t)n * 2048 + s] = (_Float16)(av[r] + biv);
    }
  }
}

// ---------------------------------------------------------------------------
// Kernel 2: flash attention per (b,h, 128-query tile). 256 threads = 8 waves,
// wave w owns query rows [w*16, w*16+16). K/V tiles (64 keys) staged in LDS,
// double-buffered: next-tile global loads issue before softmax, LDS commit
// after the P@V WMMAs; global_prefetch warms the tile after that.
// Softmax split across 256 threads (2 per row).
// ---------------------------------------------------------------------------
__global__ __launch_bounds__(256) void attn_kernel(
    const _Float16* __restrict__ qf, const _Float16* __restrict__ kf,
    const _Float16* __restrict__ vt, _Float16* __restrict__ of) {
  __shared__ __align__(16) float    sS[128 * 64];    // 32 KB scores
  __shared__ __align__(16) _Float16 sP[128 * 64];    // 16 KB probabilities
  __shared__ __align__(16) _Float16 sK[2 * 64 * 64]; // 16 KB K tiles (key-major)
  __shared__ __align__(16) _Float16 sV[2 * 64 * 64]; // 16 KB V tiles (dim-major)
  __shared__ float sM[128], sL[128], sA[128];
  __shared__ float sPmax[256], sPsum[256];

  const int tid = threadIdx.x, lane = tid & 31, wave = tid >> 5;
  const int bh = blockIdx.y;                 // 0..63
  const int qt = blockIdx.x;                 // 0..15 (128 queries each)
  const int b = bh >> 4, h = bh & 15;

  // per-wave Q fragments, reused for all key tiles
  const _Float16* qbase = qf + ((size_t)bh * 2048 + qt * 128 + wave * 16) * 64;
  const v16h qa0 = load_a_rm(qbase, 64, lane);
  const v16h qa1 = load_a_rm(qbase + 32, 64, lane);

  // cooperative K/V staging: thread copies 32B of K and 32B of V per tile
  const int crow = tid >> 2;                 // 0..63 (key for K, dim for V)
  const int cch  = (tid & 3) * 16;           // half offset 0/16/32/48
  const _Float16* kxsrc = kf + ((size_t)bh * 2048 + crow) * 64 + cch;   // + kt*64*64
  const _Float16* vxsrc = vt + ((size_t)bh * 64 + crow) * 2048 + cch;   // + kt*64
  _Float16* kdst = &sK[crow * 64 + cch];
  _Float16* vdst = &sV[crow * 64 + cch];

  // preload tile 0
  {
    v8h k0 = *(const v8h*)(kxsrc);
    v8h k1 = *(const v8h*)(kxsrc + 8);
    v8h v0 = *(const v8h*)(vxsrc);
    v8h v1 = *(const v8h*)(vxsrc + 8);
    *(v8h*)(kdst) = k0; *(v8h*)(kdst + 8) = k1;
    *(v8h*)(vdst) = v0; *(v8h*)(vdst + 8) = v1;
  }
  v8f o[4] = {v8f{}, v8f{}, v8f{}, v8f{}};
  if (tid < 128) { sM[tid] = -3.0e38f; sL[tid] = 0.f; }
  __syncthreads();

  for (int kt = 0; kt < 32; ++kt) {
    const int cur = (kt & 1) * 4096, nxt = 4096 - cur;

    // ---- issue next-tile global loads (latency overlaps softmax below) ----
    v8h pk0, pk1, pv0, pv1;
    if (kt < 31) {
      const _Float16* ks = kxsrc + (size_t)(kt + 1) * 64 * 64;
      const _Float16* vs = vxsrc + (size_t)(kt + 1) * 64;
      pk0 = *(const v8h*)(ks); pk1 = *(const v8h*)(ks + 8);
      pv0 = *(const v8h*)(vs); pv1 = *(const v8h*)(vs + 8);
    }
    // ---- warm caches for the tile after that (global_prefetch_b8) ----
    if (kt < 30) {
      __builtin_prefetch(kxsrc + (size_t)(kt + 2) * 64 * 64, 0, 0);
      __builtin_prefetch(vxsrc + (size_t)(kt + 2) * 64, 0, 0);
    }

    // ---- scores tile: S = (Q K^T) * 1/sqrt(64), from LDS-staged K ----
#pragma unroll
    for (int ns = 0; ns < 4; ++ns) {
      v8f acc = {};
      const _Float16* kb_ = &sK[cur + ns * 16 * 64];
      acc = WMMA_F16(qa0, load_b_kcont(kb_, 64, lane, 0),  acc);
      acc = WMMA_F16(qa1, load_b_kcont(kb_, 64, lane, 32), acc);
      const int col = ns * 16 + (lane & 15);
#pragma unroll
      for (int r = 0; r < 8; ++r)
        sS[(wave * 16 + cd_row(r, lane)) * 64 + col] = acc[r] * 0.125f;
    }
    __syncthreads();

    // ---- online softmax, 2 threads per row, 32 cols each ----
    const int srow_i = tid >> 1;
    const int sch    = (tid & 1) * 32;
    {
      const float* srow = &sS[srow_i * 64 + sch];
      float mt = -3.0e38f;
#pragma unroll
      for (int j = 0; j < 32; ++j) mt = fmaxf(mt, srow[j]);
      sPmax[tid] = mt;
    }
    __syncthreads();
    {
      const float mn = fmaxf(sM[srow_i], fmaxf(sPmax[srow_i * 2], sPmax[srow_i * 2 + 1]));
      const float* srow = &sS[srow_i * 64 + sch];
      _Float16* prow = &sP[srow_i * 64 + sch];
      float sum = 0.f;
#pragma unroll
      for (int j = 0; j < 32; ++j) {
        float p = __expf(srow[j] - mn);
        sum += p;
        prow[j] = (_Float16)p;
      }
      sPsum[tid] = sum;
    }
    __syncthreads();
    if (tid < 128) {
      const float mn = fmaxf(sM[tid], fmaxf(sPmax[tid * 2], sPmax[tid * 2 + 1]));
      const float alpha = __expf(sM[tid] - mn);
      sL[tid] = sL[tid] * alpha + sPsum[tid * 2] + sPsum[tid * 2 + 1];
      sM[tid] = mn;
      sA[tid] = alpha;
    }
    __syncthreads();

    // ---- rescale O accumulators by alpha(row) ----
    float al[8];
#pragma unroll
    for (int r = 0; r < 8; ++r) al[r] = sA[wave * 16 + cd_row(r, lane)];
#pragma unroll
    for (int ns = 0; ns < 4; ++ns)
#pragma unroll
      for (int r = 0; r < 8; ++r) o[ns][r] *= al[r];

    // ---- O += P @ V, from LDS-staged V (dim-major) ----
    const _Float16* prow = &sP[wave * 16 * 64];
    const v16h pa0 = load_a_rm(prow, 64, lane);
    const v16h pa1 = load_a_rm(prow + 32, 64, lane);
#pragma unroll
    for (int ns = 0; ns < 4; ++ns) {
      const _Float16* vb = &sV[cur + ns * 16 * 64];
      o[ns] = WMMA_F16(pa0, load_b_kcont(vb, 64, lane, 0),  o[ns]);
      o[ns] = WMMA_F16(pa1, load_b_kcont(vb, 64, lane, 32), o[ns]);
    }

    // ---- commit next tile into the other LDS buffer ----
    if (kt < 31) {
      *(v8h*)(kdst + nxt) = pk0; *(v8h*)(kdst + nxt + 8) = pk1;
      *(v8h*)(vdst + nxt) = pv0; *(v8h*)(vdst + nxt + 8) = pv1;
    }
    __syncthreads();
  }

  // ---- epilogue: O /= l, store merged-head f16 (B,S,E) ----
  float linv[8];
#pragma unroll
  for (int r = 0; r < 8; ++r) linv[r] = 1.0f / sL[wave * 16 + cd_row(r, lane)];
  const int col = lane & 15;
#pragma unroll
  for (int ns = 0; ns < 4; ++ns) {
#pragma unroll
    for (int r = 0; r < 8; ++r) {
      const int s = qt * 128 + wave * 16 + cd_row(r, lane);
      of[((size_t)(b * 2048 + s)) * 1024 + h * 64 + ns * 16 + col] =
          (_Float16)(o[ns][r] * linv[r]);
    }
  }
}

// ---------------------------------------------------------------------------
// Kernel 3: out = O @ Wo + bo   ([8192,1024] x [1024,1024], f32 accumulate)
// grid = (token tiles 128, col tiles 16); 128 threads = 4 waves x 16 rows.
// A-fragment software-pipelined one K-step ahead; B-fragments batched.
// ---------------------------------------------------------------------------
__global__ __launch_bounds__(128) void oproj_kernel(
    const _Float16* __restrict__ of, const _Float16* __restrict__ WoT,
    const float* __restrict__ bo, float* __restrict__ out) {
  const int tid = threadIdx.x, lane = tid & 31, wave = tid >> 5;
  const int mt = blockIdx.x;   // 0..127
  const int nt = blockIdx.y;   // 0..15

  const _Float16* arow = of + ((size_t)mt * 64 + wave * 16) * 1024;
  v8f acc[4] = {v8f{}, v8f{}, v8f{}, v8f{}};

  v16h a_cur = load_a_rm(arow, 1024, lane);
  for (int kc = 0; kc < 1024; kc += 32) {
    v16h a_nxt = a_cur;
    if (kc < 992) a_nxt = load_a_rm(arow + kc + 32, 1024, lane);
    v16h bfr[4];
#pragma unroll
    for (int ns = 0; ns < 4; ++ns)
      bfr[ns] = load_b_kcont(WoT + ((size_t)(nt * 64 + ns * 16)) * 1024, 1024, lane, kc);
#pragma unroll
    for (int ns = 0; ns < 4; ++ns)
      acc[ns] = WMMA_F16(a_cur, bfr[ns], acc[ns]);
    a_cur = a_nxt;
  }

  const int col = lane & 15;
#pragma unroll
  for (int ns = 0; ns < 4; ++ns) {
    const int n = nt * 64 + ns * 16 + col;
    const float bias = bo[n];
#pragma unroll
    for (int r = 0; r < 8; ++r) {
      const int t = mt * 64 + wave * 16 + cd_row(r, lane);
      out[(size_t)t * 1024 + n] = acc[ns][r] + bias;
    }
  }
}

// ---------------------------------------------------------------------------
// Host launcher
// ---------------------------------------------------------------------------
extern "C" void kernel_launch(void* const* d_in, const int* in_sizes, int n_in,
                              void* d_out, int out_size, void* d_ws, size_t ws_size,
                              hipStream_t stream) {
  (void)in_sizes; (void)n_in; (void)out_size; (void)ws_size;

  const float* x     = (const float*)d_in[0];
  const float* theta = (const float*)d_in[1];
  const float* Wq    = (const float*)d_in[2];
  const float* bq    = (const float*)d_in[3];
  const float* Wk    = (const float*)d_in[4];
  const float* bk    = (const float*)d_in[5];
  const float* Wv    = (const float*)d_in[6];
  const float* bv    = (const float*)d_in[7];
  const float* Wo    = (const float*)d_in[8];
  const float* bo    = (const float*)d_in[9];
  float* out = (float*)d_out;

  // workspace layout (f16 intermediates): 4x16MiB + 2MiB + 3x8KiB ~= 66 MiB
  char* ws = (char*)d_ws;
  _Float16* qf  = (_Float16*)(ws);                          // (B,H,S,64) f16
  _Float16* kf  = (_Float16*)(ws + ((size_t)16 << 20));     // (B,H,S,64) f16
  _Float16* vt  = (_Float16*)(ws + ((size_t)32 << 20));     // (B,H,64,S) f16
  _Float16* of  = (_Float16*)(ws + ((size_t)48 << 20));     // (B,S,E)    f16
  _Float16* WoT = (_Float16*)(ws + ((size_t)64 << 20));     // (E,E) transposed f16
  _Float16* WtQ = (_Float16*)(ws + ((size_t)66 << 20));     // (64,64) transposed f16
  _Float16* WtK = WtQ + 64 * 64;
  _Float16* WtV = WtK + 64 * 64;

  convert_wo_kernel<<<dim3(4096), dim3(256), 0, stream>>>(Wo, WoT);
  convert_w64_kernel<<<dim3(16), dim3(256), 0, stream>>>(Wq, Wk, Wv, WtQ, WtK, WtV);
  qkv_kernel<<<dim3(32, 64), dim3(128), 0, stream>>>(x, theta, WtQ,
                                                     bq, bk, bv, qf, kf, vt);
  attn_kernel<<<dim3(16, 64), dim3(256), 0, stream>>>(qf, kf, vt, of);
  oproj_kernel<<<dim3(128, 16), dim3(128), 0, stream>>>(of, WoT, bo, out);
}
